// CLAGCN_84267258347717
// MI455X (gfx1250) — matrix operand order repaired and checked by reading, not compiled, not run on hardware
//
#include <hip/hip_runtime.h>
#include <hip/hip_bf16.h>

#define NNODES 50000
#define NEDGES 800000
#define KDIM   128
#define ROWT   5   // row tiles per wave; 50000/16 = 3125 = 5*625 -> exact

typedef __attribute__((ext_vector_type(16))) _Float16 v16h;
typedef __attribute__((ext_vector_type(8)))  _Float16 v8h;
typedef __attribute__((ext_vector_type(8)))  float    v8f;

// ---------------- elementwise / utility kernels ----------------

__global__ void k_cast_f32_f16(const float* __restrict__ src, _Float16* __restrict__ dst, int n) {
  long long i = (long long)blockIdx.x * blockDim.x + threadIdx.x;
  if (i < n) dst[i] = (_Float16)src[i];
}

// dst (Mpad x K) = transpose(src (K x M)) in f16, zero-padded rows [M, Mpad)
__global__ void k_cast_transpose(const float* __restrict__ src, _Float16* __restrict__ dst,
                                 int K, int M, int Mpad) {
  int t = blockIdx.x * blockDim.x + threadIdx.x;
  if (t >= Mpad * K) return;
  int m = t / K, k = t % K;
  dst[t] = (m < M) ? (_Float16)src[(long long)k * M + m] : (_Float16)0.0f;
}

__global__ void k_fill(float* __restrict__ p, float v, int n) {
  long long i = (long long)blockIdx.x * blockDim.x + threadIdx.x;
  if (i < n) p[i] = v;
}

__global__ void k_deg_accum(const int* __restrict__ dst, float* __restrict__ deg, int e) {
  long long i = (long long)blockIdx.x * blockDim.x + threadIdx.x;
  if (i < e) atomicAdd(&deg[dst[i]], 1.0f);
}

__global__ void k_rsqrt_inplace(float* __restrict__ p, int n) {
  long long i = (long long)blockIdx.x * blockDim.x + threadIdx.x;
  if (i < n) p[i] = rsqrtf(p[i]);
}

// agg[dst] += h[src] * dinv[src]*dinv[dst]   (agg pre-initialized with bias + self-loop)
__global__ void k_agg_edges(const int* __restrict__ src, const int* __restrict__ dst,
                            const float* __restrict__ dinv, const float* __restrict__ h,
                            float* __restrict__ agg, int e, int F) {
  long long t = (long long)blockIdx.x * blockDim.x + threadIdx.x;
  if (t >= (long long)e * F) return;
  int ed = (int)(t / F), f = (int)(t % F);
  int s = src[ed], d = dst[ed];
  float nrm = dinv[s] * dinv[d];
  atomicAdd(&agg[(long long)d * F + f], h[(long long)s * F + f] * nrm);
}

// one block per column: mean / biased variance
__global__ void k_bn_stats(const float* __restrict__ x, float* __restrict__ mean,
                           float* __restrict__ var, int n, int F) {
  __shared__ float ss[256], sq[256];
  int f = blockIdx.x, t = threadIdx.x;
  float s = 0.f, q = 0.f;
  for (int i = t; i < n; i += 256) {
    float v = x[(long long)i * F + f];
    s += v; q += v * v;
  }
  ss[t] = s; sq[t] = q; __syncthreads();
  for (int o = 128; o > 0; o >>= 1) {
    if (t < o) { ss[t] += ss[t + o]; sq[t] += sq[t + o]; }
    __syncthreads();
  }
  if (t == 0) {
    float m = ss[0] / (float)n;
    mean[f] = m;
    var[f]  = sq[0] / (float)n - m * m;
  }
}

// out[i, coff+f] = relu((x[i,f]-mean)*rsqrt(var+eps)*gamma + beta)
__global__ void k_bn_relu(const float* __restrict__ x, const float* __restrict__ mean,
                          const float* __restrict__ var, const float* __restrict__ gamma,
                          const float* __restrict__ beta, float* __restrict__ out,
                          int n, int F, int ldo, int coff) {
  long long t = (long long)blockIdx.x * blockDim.x + threadIdx.x;
  if (t >= (long long)n * F) return;
  int i = (int)(t / F), f = (int)(t % F);
  float v = (x[t] - mean[f]) * rsqrtf(var[f] + 1e-5f) * gamma[f] + beta[f];
  out[(long long)i * ldo + coff + f] = fmaxf(v, 0.0f);
}

// single-block gate: w[0..1] = normalized sigmoids of last-row dot products
__global__ void k_gate(const float* __restrict__ h1, const float* __restrict__ h2,
                       const float* __restrict__ w1, const float* __restrict__ b1,
                       const float* __restrict__ w2, const float* __restrict__ b2,
                       int d, float* __restrict__ outw) {
  __shared__ float r1[128], r2[128];
  int t = threadIdx.x;
  float a = 0.f, b = 0.f;
  for (int i = t; i < d; i += 128) { a += h1[i] * w1[i]; b += h2[i] * w2[i]; }
  r1[t] = a; r2[t] = b; __syncthreads();
  for (int o = 64; o > 0; o >>= 1) {
    if (t < o) { r1[t] += r1[t + o]; r2[t] += r2[t + o]; }
    __syncthreads();
  }
  if (t == 0) {
    float s1 = 1.0f / (1.0f + expf(-(r1[0] + b1[0])));
    float s2 = 1.0f / (1.0f + expf(-(r2[0] + b2[0])));
    float tt = s1 + s2;
    outw[0] = s1 / tt;
    outw[1] = s2 / tt;
  }
}

__global__ void k_combine(const float* __restrict__ a, const float* __restrict__ b,
                          const float* __restrict__ w, float* __restrict__ out, int n) {
  long long i = (long long)blockIdx.x * blockDim.x + threadIdx.x;
  if (i < n) out[i] = w[0] * a[i] + w[1] * b[i];
}

// combine straight into f16 (next layer's GEMM A operand)
__global__ void k_combine_f16(const float* __restrict__ a, const float* __restrict__ b,
                              const float* __restrict__ w, _Float16* __restrict__ out, int n) {
  long long i = (long long)blockIdx.x * blockDim.x + threadIdx.x;
  if (i < n) out[i] = (_Float16)(w[0] * a[i] + w[1] * b[i]);
}

// ---------------- WMMA GEMM ----------------
// C(N x M) = A(N x 128, f16) * B(128 x M)  with B given pre-transposed as Bt (Mpad x 128, f16).
// 8 waves/block; each wave computes an 80x16 C strip (ROWT=5 row tiles reuse one B fragment)
// via 20 straight-line v_wmma_f32_16x16x32_f16 (no divergence: tile id is readfirstlane'd
// into SGPRs so all control flow is scalar and EXEC is all-ones at every WMMA).
// Optional fused epilogues write S = bias + dinv^2 * C (GCN self-loop + bias) per graph.
__global__ __launch_bounds__(256) void k_gemm_wmma(
    const _Float16* __restrict__ A, const _Float16* __restrict__ Bt,
    float* __restrict__ C,
    float* __restrict__ S1, const float* __restrict__ dv1, const float* __restrict__ bs1,
    float* __restrict__ S2, const float* __restrict__ dv2, const float* __restrict__ bs2,
    int nrows, int M, int Mpad) {
  const int K = KDIM;
  int mt = Mpad >> 4;
  int rg = (nrows >> 4) / ROWT;     // 625 row groups, exact
  int wid = __builtin_amdgcn_readfirstlane((int)threadIdx.x >> 5);  // scalar wave id
  int tile = (int)blockIdx.x * 8 + wid;
  if (tile >= rg * mt) return;      // scalar branch
  int tig = tile / mt;
  int tj  = tile % mt;
  int l    = threadIdx.x & 31;
  int half = l >> 4;
  int lr   = l & 15;
  int bcol = tj * 16 + lr;                       // < Mpad always
  const _Float16* Bp = Bt + (long long)bcol * K + 8 * half;
  const _Float16* Ap0 = A + ((long long)(tig * ROWT) * 16 + lr) * K + 8 * half;

  v8f acc[ROWT];
#pragma unroll
  for (int r = 0; r < ROWT; ++r) acc[r] = (v8f){};

#pragma unroll
  for (int kb = 0; kb < K; kb += 32) {
    // B 32x16 fragment: lane = column, K = kb+8h..+7 then kb+16+8h..+7 -> two v8h loads
    v8h blo = *(const v8h*)(Bp + kb);
    v8h bhi = *(const v8h*)(Bp + kb + 16);
    v16h bf = __builtin_shufflevector(blo, bhi,
               0, 1, 2, 3, 4, 5, 6, 7, 8, 9, 10, 11, 12, 13, 14, 15);
#pragma unroll
    for (int r = 0; r < ROWT; ++r) {
      const _Float16* ap = Ap0 + (long long)r * 16 * K + kb;
      v8h alo = *(const v8h*)ap;          // K = kb+8h .. +7
      v8h ahi = *(const v8h*)(ap + 16);   // K = kb+16+8h .. +7
      v16h af = __builtin_shufflevector(alo, ahi,
                 0, 1, 2, 3, 4, 5, 6, 7, 8, 9, 10, 11, 12, 13, 14, 15);
      acc[r] = __builtin_amdgcn_wmma_f32_16x16x32_f16(false, af, false, bf,
                                                      (short)0, acc[r], false, false);
    }
  }

  int n = tj * 16 + lr;
  bool nok = n < M;
#pragma unroll
  for (int r = 0; r < ROWT; ++r) {
#pragma unroll
    for (int q = 0; q < 8; ++q) {
      int m = (tig * ROWT + r) * 16 + q + 8 * half;
      if (nok) {
        float v = acc[r][q];
        long long idx = (long long)m * M + n;
        C[idx] = v;
        if (S1) { float d = dv1[m]; S1[idx] = bs1[n] + v * d * d; }
        if (S2) { float d = dv2[m]; S2[idx] = bs2[n] + v * d * d; }
      }
    }
  }
}

// ---------------- host orchestration ----------------

extern "C" void kernel_launch(void* const* d_in, const int* in_sizes, int n_in,
                              void* d_out, int out_size, void* d_ws, size_t ws_size,
                              hipStream_t stream) {
  const int N = NNODES, E = NEDGES, NC = 40, NCP = 48;

  const float* x1a = (const float*)d_in[0];
  const float* x1b = (const float*)d_in[1];
  const float* x2a = (const float*)d_in[2];
  const float* x2b = (const float*)d_in[3];
  const int*   ei1 = (const int*)d_in[4];
  const int*   ei2 = (const int*)d_in[5];
  const float* Wi  = (const float*)d_in[6];
  const float* bi  = (const float*)d_in[7];
  const float* gi  = (const float*)d_in[8];
  const float* bei = (const float*)d_in[9];
  const float* Wm  = (const float*)d_in[10];
  const float* bm  = (const float*)d_in[11];
  const float* gm  = (const float*)d_in[12];
  const float* bem = (const float*)d_in[13];
  const float* Wf  = (const float*)d_in[14];
  const float* bf  = (const float*)d_in[15];
  const float* fc1w1_W = (const float*)d_in[16];
  const float* fc1w1_b = (const float*)d_in[17];
  const float* fc1w2_W = (const float*)d_in[18];
  const float* fc1w2_b = (const float*)d_in[19];
  const float* aws_w1_W = (const float*)d_in[20];
  const float* aws_w1_b = (const float*)d_in[21];
  const float* aws_w2_W = (const float*)d_in[22];
  const float* aws_w2_b = (const float*)d_in[23];
  const float* fcw1_W = (const float*)d_in[24];
  const float* fcw1_b = (const float*)d_in[25];
  const float* fcw2_W = (const float*)d_in[26];
  const float* fcw2_b = (const float*)d_in[27];

  // workspace layout (floats; every slice keeps 32B alignment)
  float* wsf   = (float*)d_ws;
  float* dinv1 = wsf;                 wsf += N;
  float* dinv2 = wsf;                 wsf += N;
  float* mean  = wsf;                 wsf += 128;
  float* var   = wsf;                 wsf += 128;
  float* gw    = wsf;                 wsf += 8;
  float* h     = wsf;                 wsf += (size_t)N * 128;  // projection output
  float* a1    = wsf;                 wsf += (size_t)N * 128;  // graph-1 aggregation
  float* a2    = wsf;                 wsf += (size_t)N * 128;  // graph-2 aggregation
  float* h1    = wsf;                 wsf += (size_t)N * 128;  // branch-1 activations
  float* h2    = wsf;                 wsf += (size_t)N * 128;  // branch-2 activations
  _Float16* xh  = (_Float16*)wsf;     wsf += (size_t)N * 64;   // N*128 f16 (GEMM A)
  _Float16* wihT = (_Float16*)wsf;                             // 2 x (64 x 128)
  _Float16* wmhT = wihT + 2 * 64 * 128;                        // 2 x (128 x 128)
  _Float16* wfhT = wmhT + 2 * 128 * 128;                       // 48 x 128 (zero-padded)

  dim3 b256(256);
  auto g1d = [](long long n) { return dim3((unsigned)((n + 255) / 256)); };
  auto gemm = [&](const _Float16* Ah, const _Float16* BTh, float* Cc, int M, int Mpad,
                  float* S1, const float* dv1, const float* bs1,
                  float* S2, const float* dv2, const float* bs2) {
    int tiles = ((N / 16) / ROWT) * (Mpad / 16);
    k_gemm_wmma<<<dim3((unsigned)((tiles + 7) / 8)), b256, 0, stream>>>(
        Ah, BTh, Cc, S1, dv1, bs1, S2, dv2, bs2, N, M, Mpad);
  };

  // ---- GCN symmetric norms (deg includes self-loop) ----
  k_fill<<<g1d(N), b256, 0, stream>>>(dinv1, 1.0f, N);
  k_fill<<<g1d(N), b256, 0, stream>>>(dinv2, 1.0f, N);
  k_deg_accum<<<g1d(E), b256, 0, stream>>>(ei1 + E, dinv1, E);
  k_deg_accum<<<g1d(E), b256, 0, stream>>>(ei2 + E, dinv2, E);
  k_rsqrt_inplace<<<g1d(N), b256, 0, stream>>>(dinv1, N);
  k_rsqrt_inplace<<<g1d(N), b256, 0, stream>>>(dinv2, N);

  // ---- transpose+cast all weights to f16 once (zero-padded columns) ----
  for (int i = 0; i < 2; ++i)
    k_cast_transpose<<<g1d(64 * 128), b256, 0, stream>>>(Wi + i * 128 * 64,
                                                         wihT + i * 64 * 128, 128, 64, 64);
  for (int L = 0; L < 2; ++L)
    k_cast_transpose<<<g1d(128 * 128), b256, 0, stream>>>(Wm + L * 128 * 128,
                                                          wmhT + L * 128 * 128, 128, 128, 128);
  k_cast_transpose<<<g1d(NCP * 128), b256, 0, stream>>>(Wf, wfhT, 128, NC, NCP);

  const float* xin[2][2] = {{x1a, x1b}, {x2a, x2b}};
  const int*   srcs[2]  = {ei1, ei2};
  const int*   dsts[2]  = {ei1 + E, ei2 + E};
  float*       dinvs[2] = {dinv1, dinv2};
  float*       htgt[2]  = {h1, h2};

  // ---- input layer: 2 graphs x 2 concat parts (NHID=64 each) ----
  for (int g = 0; g < 2; ++g) {
    for (int i = 0; i < 2; ++i) {
      k_cast_f32_f16<<<g1d((long long)N * 128), b256, 0, stream>>>(xin[g][i], xh, N * 128);
      gemm(xh, wihT + i * 64 * 128, h, 64, 64,
           a1, dinvs[g], bi + i * 64, nullptr, nullptr, nullptr);
      k_agg_edges<<<g1d((long long)E * 64), b256, 0, stream>>>(srcs[g], dsts[g], dinvs[g],
                                                               h, a1, E, 64);
      k_bn_stats<<<dim3(64), b256, 0, stream>>>(a1, mean, var, N, 64);
      k_bn_relu<<<g1d((long long)N * 64), b256, 0, stream>>>(a1, mean, var, gi + i * 64,
                                                             bei + i * 64, htgt[g], N, 64, 128, i * 64);
    }
  }
  k_gate<<<dim3(1), dim3(128), 0, stream>>>(h1 + (size_t)(N - 1) * 128, h2 + (size_t)(N - 1) * 128,
                                            fc1w1_W, fc1w1_b, fc1w2_W, fc1w2_b, 128, gw);
  k_combine_f16<<<g1d((long long)N * 128), b256, 0, stream>>>(h1, h2, gw, xh, N * 128);

  // ---- middle layers: h = x@Wm shared between both graphs; self-loop init fused ----
  for (int L = 0; L < 2; ++L) {
    gemm(xh, wmhT + L * 128 * 128, h, 128, 128,
         a1, dinv1, bm + L * 128, a2, dinv2, bm + L * 128);
    k_agg_edges<<<g1d((long long)E * 128), b256, 0, stream>>>(ei1, ei1 + E, dinv1, h, a1, E, 128);
    k_agg_edges<<<g1d((long long)E * 128), b256, 0, stream>>>(ei2, ei2 + E, dinv2, h, a2, E, 128);
    k_bn_stats<<<dim3(128), b256, 0, stream>>>(a1, mean, var, N, 128);
    k_bn_relu<<<g1d((long long)N * 128), b256, 0, stream>>>(a1, mean, var, gm + L * 128,
                                                            bem + L * 128, a1, N, 128, 128, 0);
    k_bn_stats<<<dim3(128), b256, 0, stream>>>(a2, mean, var, N, 128);
    k_bn_relu<<<g1d((long long)N * 128), b256, 0, stream>>>(a2, mean, var, gm + L * 128,
                                                            bem + L * 128, a2, N, 128, 128, 0);
    k_gate<<<dim3(1), dim3(128), 0, stream>>>(a1 + (size_t)(N - 1) * 128, a2 + (size_t)(N - 1) * 128,
                                              aws_w1_W + L * 128, aws_w1_b + L,
                                              aws_w2_W + L * 128, aws_w2_b + L, 128, gw);
    k_combine_f16<<<g1d((long long)N * 128), b256, 0, stream>>>(a1, a2, gw, xh, N * 128);
  }

  // ---- final layer: p1/p2 self-loop init fused into GEMM, aggregated in d_out ----
  float* outp = (float*)d_out;
  float* p1 = outp + (size_t)N * NC;
  float* p2 = outp + (size_t)2 * N * NC;
  gemm(xh, wfhT, h, NC, NCP, p1, dinv1, bf, p2, dinv2, bf);  // h reused as N x 40
  k_agg_edges<<<g1d((long long)E * NC), b256, 0, stream>>>(ei1, ei1 + E, dinv1, h, p1, E, NC);
  k_agg_edges<<<g1d((long long)E * NC), b256, 0, stream>>>(ei2, ei2 + E, dinv2, h, p2, E, NC);
  k_gate<<<dim3(1), dim3(128), 0, stream>>>(p1 + (size_t)(N - 1) * NC, p2 + (size_t)(N - 1) * NC,
                                            fcw1_W, fcw1_b, fcw2_W, fcw2_b, NC, gw);
  k_combine<<<g1d((long long)N * NC), b256, 0, stream>>>(p1, p2, gw, outp, N * NC);

  (void)in_sizes; (void)n_in; (void)out_size; (void)ws_size;
}
